// FusedSelfAttention_25744033972242
// MI455X (gfx1250) — matrix operand order repaired and checked by reading, not compiled, not run on hardware
//
#include <hip/hip_runtime.h>
#include <cstdint>
#include <cstddef>

// ---------------------------------------------------------------------------
// Fused self-attention for MI455X (gfx1250), wave32 + WMMA f16 (f32 acc).
// B=2, T=2048, C=1024, H=16, HD=64.
// Pipeline: cvt(f32->f16) -> QKV GEMM (wmma, 128x128 tile, async LDS fills) ->
//           flash attention (S^T trick, wmma, async K fill) ->
//           out GEMM (wmma, f32 out).
// ---------------------------------------------------------------------------

#define ATT_B   2
#define ATT_T   2048
#define ATT_C   1024
#define ATT_H   16
#define ATT_HD  64
#define ATT_3C  3072
#define ATT_BT  (ATT_B * ATT_T)      // 4096
#define NEGBIG  (-1.0e30f)

typedef __attribute__((ext_vector_type(16))) _Float16 v16h;
typedef __attribute__((ext_vector_type(8)))  float    v8f;

union FragU { v16h h; uint32_t u[8]; };
union H8   { uint4 v; _Float16 h[8]; };

// --- CDNA5 async global->LDS copy (ASYNCcnt path; bypasses VGPRs) ----------
// LDS byte offset = low 32 bits of the generic pointer
__device__ inline void async_load_b128(void* lds_dst, const void* gsrc) {
  const uint32_t loff = (uint32_t)(uintptr_t)lds_dst;
  const uint64_t gadr = (uint64_t)(uintptr_t)gsrc;
  asm volatile("global_load_async_to_lds_b128 %0, %1, off"
               :: "v"(loff), "v"(gadr)
               : "memory");
}
__device__ inline void wait_async0() {
#if __has_builtin(__builtin_amdgcn_s_wait_asynccnt)
  __builtin_amdgcn_s_wait_asynccnt(0);
#else
  asm volatile("s_wait_asynccnt 0x0" ::: "memory");
#endif
}

// A-fragment, 16(M) x 32(K) f16, from [row][K] storage with K contiguous.
// Lane L: M = L%16, grp = L/16. VGPR v pairs: K = grp*8+2v (v<4) else 16+grp*8+2(v-4).
__device__ inline v16h load_a_frag(const _Float16* base, int stride) {
  const int lane = threadIdx.x & 31;
  const int mr = lane & 15, grp = lane >> 4;
  const _Float16* rp = base + (size_t)mr * stride;
  FragU f;
#pragma unroll
  for (int v = 0; v < 8; ++v) {
    const int k = (v < 4) ? (grp * 8 + 2 * v) : (16 + grp * 8 + 2 * (v - 4));
    f.u[v] = *(const uint32_t*)(rp + k);
  }
  return f.h;
}

// B-fragment, 32(K) x 16(N) f16, from [N][K] storage with K contiguous.
// Lane L: N = L%16, grp = L/16. VGPR v pairs: K = grp*8+2v (v<4) else 16+grp*8+2(v-4).
__device__ inline v16h load_b_frag(const _Float16* base, int stride) {
  const int lane = threadIdx.x & 31;
  const int nr = lane & 15, grp = lane >> 4;
  const _Float16* rp = base + (size_t)nr * stride;
  FragU f;
#pragma unroll
  for (int v = 0; v < 8; ++v) f.u[v] = *(const uint32_t*)(rp + ((v < 4) ? (grp * 8 + 2 * v) : (16 + grp * 8 + 2 * (v - 4))));
  return f.h;
}

__device__ inline v8f wmma_f16(v16h a, v16h b, v8f c) {
  return __builtin_amdgcn_wmma_f32_16x16x32_f16(false, a, false, b, (short)0, c,
                                                false, false);
}

// ---------------------------------------------------------------------------
// f32 -> f16 convert
// ---------------------------------------------------------------------------
__global__ void cvt_f32_to_f16_kernel(const float* __restrict__ src,
                                      _Float16* __restrict__ dst, int n) {
  for (int i = blockIdx.x * blockDim.x + threadIdx.x; i < n;
       i += gridDim.x * blockDim.x)
    dst[i] = (_Float16)src[i];
}

// ---------------------------------------------------------------------------
// GEMM: C[M,N] = A[M,K](f16 rowmaj) * Bw[K,N](f16 rowmaj) + bias[N](f32)
// Block tile 128x128, 256 threads = 8 waves; wave w -> 32 rows x 64 cols
// (2 M-tiles x 4 N-tiles = 8 WMMA per k-step, 4x fragment reuse).
// A tile filled with async global->LDS b128 copies (no VGPR round trip);
// B tile transposed through VGPRs (layout change requires it).
// ---------------------------------------------------------------------------
template <bool F32OUT>
__global__ __launch_bounds__(256) void gemm_bias_kernel(
    const _Float16* __restrict__ A, const _Float16* __restrict__ Bw,
    const float* __restrict__ bias, void* __restrict__ Cout,
    int M, int N, int K) {
  __shared__ _Float16 Alds[128 * 32];   // [M][K]
  __shared__ _Float16 Blds[128 * 32];   // [N][K] (transposed on fill)

  const int tid  = threadIdx.x;
  const int wave = tid >> 5;
  const int m0 = blockIdx.y * 128, n0 = blockIdx.x * 128;
  const int mw = (wave >> 1) * 32;   // wave M offset in block: 0/32/64/96
  const int nw = (wave & 1) * 64;    // wave N offset in block: 0/64

  v8f acc[2][4];
#pragma unroll
  for (int mi = 0; mi < 2; ++mi)
#pragma unroll
    for (int ni = 0; ni < 4; ++ni)
#pragma unroll
      for (int r = 0; r < 8; ++r) acc[mi][ni][r] = 0.f;

  for (int kb = 0; kb < K; kb += 32) {
    __syncthreads();
    {   // A tile: async copy, thread -> (m, 16 halves)
      const int m = tid >> 1, kq = (tid & 1) * 16;
      const _Float16* src = A + (size_t)(m0 + m) * K + kb + kq;
      _Float16* dst = Alds + m * 32 + kq;
      async_load_b128(dst, src);
      async_load_b128(dst + 8, src + 8);
    }
    {   // B tile, transposed into [N][K]
      const int k = tid >> 3, nq = (tid & 7) * 16;
      const _Float16* src = Bw + (size_t)(kb + k) * N + n0 + nq;
      H8 d0; d0.v = *(const uint4*)(src);
      H8 d1; d1.v = *(const uint4*)(src + 8);
#pragma unroll
      for (int i = 0; i < 8; ++i) {
        Blds[(nq + i) * 32 + k]     = d0.h[i];
        Blds[(nq + 8 + i) * 32 + k] = d1.h[i];
      }
      if (kb + 32 < K) __builtin_prefetch(src + (size_t)32 * N, 0, 1);
    }
    wait_async0();          // async fills must complete before the barrier
    __syncthreads();

    v16h af[2];
#pragma unroll
    for (int mi = 0; mi < 2; ++mi)
      af[mi] = load_a_frag(Alds + (mw + mi * 16) * 32, 32);
#pragma unroll
    for (int ni = 0; ni < 4; ++ni) {
      const v16h bf = load_b_frag(Blds + (nw + ni * 16) * 32, 32);
#pragma unroll
      for (int mi = 0; mi < 2; ++mi)
        acc[mi][ni] = wmma_f16(af[mi], bf, acc[mi][ni]);
    }
  }

  // Epilogue (C layout: lane holds N=lane%16, rows r + 8*(lane/16))
  const int lane = tid & 31, nc = lane & 15, grp = lane >> 4;
#pragma unroll
  for (int mi = 0; mi < 2; ++mi)
#pragma unroll
    for (int ni = 0; ni < 4; ++ni) {
      const int ng = n0 + nw + ni * 16 + nc;
      const float bv = bias[ng];
#pragma unroll
      for (int r = 0; r < 8; ++r) {
        const int mg = m0 + mw + mi * 16 + r + 8 * grp;
        const float val = acc[mi][ni][r] + bv;
        if (F32OUT)
          ((float*)Cout)[(size_t)mg * N + ng] = val;
        else
          ((_Float16*)Cout)[(size_t)mg * N + ng] = (_Float16)val;
      }
    }
}

// ---------------------------------------------------------------------------
// Flash attention. Grid (T/64, H, B), 128 threads = 4 waves.
// Wave w owns q rows [qblk+16w, +16). Computes S^T = K*Q^T so that each lane
// holds 16 scores of ONE q column in registers: softmax reductions become
// in-register + a single xor-16 shuffle. K tile filled via async global->LDS.
// qkv layout: [(b*T+t)*3072 + s*1024 + h*64 + d], s in {q,k,v}.
// ---------------------------------------------------------------------------
__global__ __launch_bounds__(128) void attention_kernel(
    const _Float16* __restrict__ qkv, const int* __restrict__ kpm,
    _Float16* __restrict__ outh) {
  __shared__ _Float16 Klds[32 * 64];          // [key][d]
  __shared__ _Float16 Vlds[64 * 32];          // [d][key]
  __shared__ float    Mlds[32];               // additive key-padding mask
  __shared__ _Float16 Plds[4 * 16 * 34];      // per-wave P, stride 34 (no bank conflicts)

  const int tid = threadIdx.x, wave = tid >> 5, lane = tid & 31;
  const int nc = lane & 15, grp = lane >> 4;
  const int qblk = blockIdx.x * 64;
  const int h = blockIdx.y, b = blockIdx.z;
  const int qrow0 = qblk + wave * 16;
  const size_t rowbase = (size_t)(b * ATT_T) * ATT_3C + h * ATT_HD;

  // Q^T as B-fragments (N=q, K=d): natural [q][d] storage, loaded once.
  const _Float16* Qbase = qkv + rowbase + (size_t)qrow0 * ATT_3C;
  const v16h bq0 = load_b_frag(Qbase, ATT_3C);
  const v16h bq1 = load_b_frag(Qbase + 32, ATT_3C);

  v8f o[4];                       // O: rows q (r+8grp), cols d (t*16+nc)
#pragma unroll
  for (int t = 0; t < 4; ++t)
#pragma unroll
    for (int r = 0; r < 8; ++r) o[t][r] = 0.f;
  float m_ = NEGBIG, l_ = 0.f;    // per-lane stats for q column qrow0 + nc

  const int kend_wave = qrow0 + 16;   // exclusive causal bound for this wave
  const int kend_blk  = qblk + 64;    // all waves run full block loop (barriers)
  const int kpm_base  = b * ATT_T;

  for (int kt = 0; kt < kend_blk; kt += 32) {
    __syncthreads();
    {   // cooperative K/V tile load: thread -> key kk, 16 d values
      const int kk = tid >> 2, dq = (tid & 3) * 16;
      // K tile: contiguous copy -> async global->LDS b128 (ASYNCcnt path)
      const _Float16* krow = qkv + rowbase + (size_t)(kt + kk) * ATT_3C + 1024 + dq;
      _Float16* kdst = Klds + kk * 64 + dq;
      async_load_b128(kdst, krow);
      async_load_b128(kdst + 8, krow + 8);
      // V tile: transposed store [d][key] (needs VGPR round trip)
      const _Float16* vrow = qkv + rowbase + (size_t)(kt + kk) * ATT_3C + 2048 + dq;
      H8 v0; v0.v = *(const uint4*)(vrow);
      H8 v1; v1.v = *(const uint4*)(vrow + 8);
#pragma unroll
      for (int i = 0; i < 8; ++i) {
        Vlds[(dq + i)     * 32 + kk] = v0.h[i];
        Vlds[(dq + 8 + i) * 32 + kk] = v1.h[i];
      }
      if (tid < 32) Mlds[tid] = (kpm[kpm_base + kt + tid] != 0) ? NEGBIG : 0.0f;
    }
    wait_async0();
    __syncthreads();

    if (kt < kend_wave) {
      // S^T = K Q^T: two 16-key subtiles (A frags from Klds [key][d])
      v8f st[2];
#pragma unroll
      for (int j = 0; j < 2; ++j) {
#pragma unroll
        for (int r = 0; r < 8; ++r) st[j][r] = 0.f;
        const v16h ak0 = load_a_frag(Klds + (16 * j) * 64, 64);
        const v16h ak1 = load_a_frag(Klds + (16 * j) * 64 + 32, 64);
        st[j] = wmma_f16(ak0, bq0, st[j]);
        st[j] = wmma_f16(ak1, bq1, st[j]);
      }

      // scale + masks; in-register max over this lane's 16 key-scores
      const int q = qrow0 + nc;
      float sv[2][8];
      float mx = NEGBIG;
#pragma unroll
      for (int j = 0; j < 2; ++j)
#pragma unroll
        for (int r = 0; r < 8; ++r) {
          const int kl = 16 * j + r + 8 * grp;        // key local index
          float v = st[j][r] * 0.125f + Mlds[kl];     // 1/sqrt(HD) + pad mask
          if (kt + kl > q) v = NEGBIG;                // causal
          sv[j][r] = v;
          mx = fmaxf(mx, v);
        }
      mx = fmaxf(mx, __shfl_xor(mx, 16, 32));         // other half of keys
      const float mnew  = fmaxf(m_, mx);
      const float alpha = __expf(m_ - mnew);

      // P^T -> Plds as [q][key] (stride 34) while accumulating row sum
      _Float16* Pw = Plds + wave * (16 * 34);
      float rs = 0.f;
#pragma unroll
      for (int j = 0; j < 2; ++j)
#pragma unroll
        for (int r = 0; r < 8; ++r) {
          const float e = __expf(sv[j][r] - mnew);
          rs += e;
          Pw[nc * 34 + 16 * j + r + 8 * grp] = (_Float16)e;
        }
      rs += __shfl_xor(rs, 16, 32);
      l_ = l_ * alpha + rs;
      m_ = mnew;

      // rescale O rows by alpha of their q (broadcast from lane nc==q_local)
#pragma unroll
      for (int r = 0; r < 8; ++r) {
        const float a = __shfl(alpha, r + 8 * grp, 32);
#pragma unroll
        for (int t = 0; t < 4; ++t) o[t][r] *= a;
      }

      // O += P V  (A frag = P from Plds [q][key]; Vlds is [N=d][K=key])
      const v16h ap = load_a_frag(Pw, 34);
#pragma unroll
      for (int t = 0; t < 4; ++t) {
        const v16h bv = load_b_frag(Vlds + (t * 16) * 32, 32);
        o[t] = wmma_f16(ap, bv, o[t]);
      }
    }
  }

  // normalize + store f16 into attn_out [B*T, C] at column h*64+d
  const float inv = (l_ > 0.f) ? (1.0f / l_) : 0.f;
#pragma unroll
  for (int r = 0; r < 8; ++r) {
    const float ir = __shfl(inv, r + 8 * grp, 32);
    const int row = qrow0 + r + 8 * grp;
#pragma unroll
    for (int t = 0; t < 4; ++t) {
      const int d = t * 16 + nc;
      outh[(size_t)(b * ATT_T + row) * ATT_C + h * ATT_HD + d] =
          (_Float16)(o[t][r] * ir);
    }
  }
}

// ---------------------------------------------------------------------------
// launch
// ---------------------------------------------------------------------------
extern "C" void kernel_launch(void* const* d_in, const int* in_sizes, int n_in,
                              void* d_out, int out_size, void* d_ws,
                              size_t ws_size, hipStream_t stream) {
  (void)in_sizes; (void)n_in; (void)out_size; (void)ws_size;
  const float* x    = (const float*)d_in[0];
  // d_in[1] = attn_mask (pure causal; applied analytically in-kernel)
  const int*   kpm  = (const int*)d_in[2];
  const float* Wqkv = (const float*)d_in[3];
  const float* bqkv = (const float*)d_in[4];
  const float* Wout = (const float*)d_in[5];
  const float* bout = (const float*)d_in[6];

  char* ws = (char*)d_ws;   // 48 MiB used
  _Float16* Xh    = (_Float16*)(ws);                          // 8 MiB
  _Float16* Wqkvh = (_Float16*)(ws + ((size_t)8  << 20));     // 6 MiB
  _Float16* Wouth = (_Float16*)(ws + ((size_t)14 << 20));     // 2 MiB
  _Float16* QKVh  = (_Float16*)(ws + ((size_t)16 << 20));     // 24 MiB
  _Float16* AttnH = (_Float16*)(ws + ((size_t)40 << 20));     // 8 MiB

  cvt_f32_to_f16_kernel<<<1024, 256, 0, stream>>>(x, Xh, ATT_BT * ATT_C);
  cvt_f32_to_f16_kernel<<<1024, 256, 0, stream>>>(Wqkv, Wqkvh, ATT_C * ATT_3C);
  cvt_f32_to_f16_kernel<<<1024, 256, 0, stream>>>(Wout, Wouth, ATT_C * ATT_C);

  gemm_bias_kernel<false><<<dim3(ATT_3C / 128, ATT_BT / 128), 256, 0, stream>>>(
      Xh, Wqkvh, bqkv, (void*)QKVh, ATT_BT, ATT_3C, ATT_C);

  attention_kernel<<<dim3(ATT_T / 64, ATT_H, ATT_B), 128, 0, stream>>>(
      QKVh, kpm, AttnH);

  gemm_bias_kernel<true><<<dim3(ATT_C / 128, ATT_BT / 128), 256, 0, stream>>>(
      AttnH, Wouth, bout, d_out, ATT_BT, ATT_C, ATT_C);
}
